// TNModule_54829552501061
// MI455X (gfx1250) — matrix operand order repaired and checked by reading, not compiled, not run on hardware
//
#include <hip/hip_runtime.h>
#include <hip/hip_bf16.h>

// Problem constants (from reference)
#define NTT 4096   // NUM_NODES + SEQ_LEN
#define BB  4
#define F   32
#define WAVES_PER_BLOCK 4

typedef __attribute__((ext_vector_type(16))) __bf16 v16bf;
typedef __attribute__((ext_vector_type(8)))  float  v8f;

struct __align__(16) U4 { unsigned int x, y, z, w; };

union BfVec {
    v16bf v;
    U4 q[2];
    __bf16 h[16];
};

__device__ __forceinline__ float fast_tanh(float x) {
#if __has_builtin(__builtin_amdgcn_tanhf)
    return __builtin_amdgcn_tanhf(x);  // v_tanh_f32 (gfx1250 trans op)
#else
    return tanhf(x);
#endif
}

__device__ __forceinline__ float clamp01(float x) {
    // med3(x,0,1): backend folds this into the clamp modifier of the
    // producing instruction when possible (else one v_med3_f32).
    return __builtin_amdgcn_fmed3f(x, 0.0f, 1.0f);
}

// ---- X (f32) -> Xbf (bf16), done once per launch ----
__global__ void xcvt_kernel(const float* __restrict__ X,
                            __bf16* __restrict__ Xbf, int n) {
    int i = blockIdx.x * blockDim.x + threadIdx.x;
    if (i < n) Xbf[i] = (__bf16)X[i];
}

// ---- P^T[b][g][m] = bf16( sum_f H[b][m][f] * W[f][g] )  (tiny matmul) ----
__global__ void pt_kernel(const float* __restrict__ H,
                          const float* __restrict__ W,
                          __bf16* __restrict__ Pt) {
    int t = blockIdx.x * blockDim.x + threadIdx.x;   // b*32*NTT + g*NTT + m
    int m = t & (NTT - 1);
    int g = (t >> 12) & 31;
    int b = t >> 17;
    const float* h = H + ((size_t)(b * NTT + m)) * F;
    float s = 0.f;
#pragma unroll
    for (int f = 0; f < F; ++f) s = fmaf(h[f], W[f * F + g], s);  // W uniform -> s_load
    Pt[t] = (__bf16)s;
}

// ---- fused: out[b,n,:] = elu( sum_m tanh(relu(x_n . x_m) + I) * P[m,:] ) ----
// simT trick: compute simT[m][n] = x_m . x_n  (A = X m-rows, B = X^T n-cols).
// Its C layout (lane = n, VGPR r = m-within-subtile) is EXACTLY the A-operand
// layout needed by the aggregation WMMA (lane = n, K = m): subtile0 r=0..7 ->
// K elems 0..7, subtile1 r=0..7 -> K elems 8..15. adj is symmetric, so the
// values are the right ones -- zero cross-lane movement, no LDS staging.
// Nonlinearity: tanh(relu(x)) == clamp01(tanh(x)) (tanh monotone, tanh(0)=0,
// tanh<=1), so relu costs zero extra instructions in the hot path.
__global__ void __launch_bounds__(WAVES_PER_BLOCK * 32)
gcn_kernel(const __bf16* __restrict__ Xbf,
           const __bf16* __restrict__ Pt,
           float* __restrict__ Out) {
    const int lane     = threadIdx.x & 31;
    const int wave     = threadIdx.x >> 5;
    // wave-uniform scalars -> SGPRs (SALU tile math, scalar diag branch)
    const int widU     = __builtin_amdgcn_readfirstlane(blockIdx.x * WAVES_PER_BLOCK + wave);
    const int b        = widU >> 8;              // NTT/16 = 256 row-tiles per batch
    const int n0       = (widU & 255) << 4;
    const int lane16   = lane & 15;
    const int laneHalf = lane >> 4;
    const int mlocBase = laneHalf * 8;

    // Loop-invariant B operand (32x16 bf16): K = f, N = n -> column n comes from
    // row n0+lane16 of X; lanes 0-15 need f={0..7,16..23}, lanes 16-31 need f={8..15,24..31}.
    const U4* nrow = (const U4*)(Xbf + ((size_t)(b * NTT + n0 + lane16)) * F);
    BfVec bN;
    bN.q[0] = nrow[laneHalf];
    bN.q[1] = nrow[2 + laneHalf];

    v8f acc0 = {};   // output features g = 0..15   (C: lane = g, VGPR r = nloc)
    v8f acc1 = {};   // output features g = 16..31

    for (int m0 = 0; m0 < NTT; m0 += 32) {
        // ---- both sim subtiles back-to-back (independent WMMAs) ----------
        // A operand (16x32 bf16): M = m-row = lane16, K = f
        // laneHalf=0: f = {0..7, 16..23}; laneHalf=1: f = {8..15, 24..31}
        const U4* mrow0 = (const U4*)(Xbf + ((size_t)(b * NTT + m0 + lane16)) * F);
        const U4* mrow1 = (const U4*)(Xbf + ((size_t)(b * NTT + m0 + 16 + lane16)) * F);
        BfVec aM0, aM1;
        aM0.q[0] = mrow0[laneHalf];
        aM0.q[1] = mrow0[2 + laneHalf];
        aM1.q[0] = mrow1[laneHalf];
        aM1.q[1] = mrow1[2 + laneHalf];
        v8f zero = {};
        v8f sim0 = __builtin_amdgcn_wmma_f32_16x16x32_bf16(
            false, aM0.v, false, bN.v, (short)0, zero, false, false);
        v8f sim1 = __builtin_amdgcn_wmma_f32_16x16x32_bf16(
            false, aM1.v, false, bN.v, (short)0, zero, false, false);

        // ---- nonlinearity, built in-lane into the next A operand ---------
        BfVec a2;   // adj tile as A operand: lane = n, K = m
        if ((unsigned)(n0 - m0) < 32u) {
            // scalar branch, taken for 1 of 128 chunks: chunk contains the
            // diagonal; must add +1 BEFORE tanh where m == n.
#pragma unroll
            for (int s = 0; s < 2; ++s) {
                const v8f sim = s ? sim1 : sim0;
                const int mbase = m0 + s * 16;
#pragma unroll
                for (int r = 0; r < 8; ++r) {
                    float v = fmaxf(sim[r], 0.0f);
                    if ((mbase + mlocBase + r) == (n0 + lane16)) v += 1.0f;
                    a2.h[s * 8 + r] = (__bf16)fast_tanh(v);
                }
            }
        } else {
#pragma unroll
            for (int r = 0; r < 8; ++r)
                a2.h[r] = (__bf16)clamp01(fast_tanh(sim0[r]));
#pragma unroll
            for (int r = 0; r < 8; ++r)
                a2.h[8 + r] = (__bf16)clamp01(fast_tanh(sim1[r]));
        }

        // ---- aggregation: acc[g-half] += adj_tile(16n x 32m) @ P^T -------
        {
            const U4* p0 = (const U4*)(Pt + ((size_t)(b * 32 + lane16)) * NTT
                                           + m0 + laneHalf * 8);
            BfVec b2;
            b2.q[0] = p0[0];
            b2.q[1] = p0[2];
            acc0 = __builtin_amdgcn_wmma_f32_16x16x32_bf16(
                false, a2.v, false, b2.v, (short)0, acc0, false, false);
            const U4* p1 = (const U4*)(Pt + ((size_t)(b * 32 + 16 + lane16)) * NTT
                                           + m0 + laneHalf * 8);
            b2.q[0] = p1[0];
            b2.q[1] = p1[2];
            acc1 = __builtin_amdgcn_wmma_f32_16x16x32_bf16(
                false, a2.v, false, b2.v, (short)0, acc1, false, false);
        }
    }

    // ---- ELU (alpha=1) + store f32 output tile ----
#pragma unroll
    for (int r = 0; r < 8; ++r) {
        const int nloc = (laneHalf * 8) + r;
        float v0 = acc0[r];
        float v1 = acc1[r];
        v0 = v0 > 0.f ? v0 : expm1f(v0);
        v1 = v1 > 0.f ? v1 : expm1f(v1);
        float* orow = Out + ((size_t)(b * NTT + n0 + nloc)) * F;
        orow[lane16]      = v0;   // C layout: N = lane16 (acc0 -> g, acc1 -> g+16)
        orow[16 + lane16] = v1;
    }
}

extern "C" void kernel_launch(void* const* d_in, const int* in_sizes, int n_in,
                              void* d_out, int out_size, void* d_ws, size_t ws_size,
                              hipStream_t stream) {
    (void)in_sizes; (void)n_in; (void)out_size; (void)ws_size;
    const float* X = (const float*)d_in[0];   // (B, NT, 32) f32
    const float* W = (const float*)d_in[1];   // (2, 32, 32) f32
    float* Out = (float*)d_out;               // (B, NT, 32) f32

    char* ws = (char*)d_ws;
    __bf16* Xbf = (__bf16*)ws;                                  // 1 MB
    __bf16* Pt  = (__bf16*)(ws + (size_t)BB * NTT * F * 2);     // 1 MB
    float*  H1  = (float*)(ws + (size_t)2 * BB * NTT * F * 2);  // 2 MB

    const int total = BB * NTT * F;                              // 524288
    const int gcnBlocks = (BB * (NTT / 16)) / WAVES_PER_BLOCK;   // 256

    xcvt_kernel<<<total / 256, 256, 0, stream>>>(X, Xbf, total);

    // layer 0: adj(X) @ X @ W0 -> H1
    pt_kernel<<<total / 256, 256, 0, stream>>>(X, W, Pt);
    gcn_kernel<<<gcnBlocks, WAVES_PER_BLOCK * 32, 0, stream>>>(Xbf, Pt, H1);

    // layer 1: adj(X) @ H1 @ W1 -> Out
    pt_kernel<<<total / 256, 256, 0, stream>>>(H1, W + F * F, Pt);
    gcn_kernel<<<gcnBlocks, WAVES_PER_BLOCK * 32, 0, stream>>>(Xbf, Pt, Out);
}
